// AugmentedConv_46076409152142
// MI455X (gfx1250) — matrix operand order repaired and checked by reading, not compiled, not run on hardware
//
#include <hip/hip_runtime.h>

// ---------------------------------------------------------------------------
// AugmentedConv (attention-augmented conv1d) for MI455X / gfx1250.
//   B=16, C_IN=128, L=1024, C_OUT=128, DK=64, NH=8 (dkh=8), DV=128 (dvh=16)
// Strategy:
//  * conv_out + qkv conv: one bf16 WMMA GEMM, M=320, K=384(im2col), N=16K.
//  * attention: flash-style fusion (never materialize the 537MB logits),
//    f32 16x16x4 WMMA for the K=8 products (QK^T, rel-logits),
//    ds_bpermute diagonal gather for rel_to_abs, bf16 16x16x32 WMMA for P*V.
//  * the reference's reshape "bug" ([L,dvh]->[dvh,L] reshape) is a flat
//    reinterpret in row-major, so the final 1x1 conv just reads the flat
//    attention buffer with a different channel view.
// ---------------------------------------------------------------------------

typedef __attribute__((ext_vector_type(2)))  float  v2f;
typedef __attribute__((ext_vector_type(8)))  float  v8f;
typedef __attribute__((ext_vector_type(8)))  __bf16 v8bf;
typedef __attribute__((ext_vector_type(16))) __bf16 v16bf;

#define N_B   16
#define C_IN  128
#define L_SEQ 1024
#define N_H   8
#define DKH_  8
#define DVH_  16

__device__ __forceinline__ v8f wmma_f32_k4(v2f a, v2f b, v8f c) {
  // V_WMMA_F32_16X16X4_F32  (K=4, f32 in / f32 out)
  return __builtin_amdgcn_wmma_f32_16x16x4_f32(false, a, false, b, (short)0, c,
                                               false, false);
}
__device__ __forceinline__ v8f wmma_bf16(v16bf a, v16bf b, v8f c) {
  // V_WMMA_F32_16X16X32_BF16
  return __builtin_amdgcn_wmma_f32_16x16x32_bf16(false, a, false, b, (short)0, c,
                                                 false, false);
}
__device__ __forceinline__ float bperm(int srclane, float v) {
  return __int_as_float(
      __builtin_amdgcn_ds_bpermute(srclane << 2, __float_as_int(v)));
}
__device__ __forceinline__ v2f mk2(float a, float b) {
  v2f r; r.x = a; r.y = b; return r;
}

// ---------------------------------------------------------------------------
// Pack conv weights [320][384] (row = out-ch, col = ci*3+k) into bf16.
// ---------------------------------------------------------------------------
__global__ void pack_w_kernel(const float* __restrict__ w_conv,
                              const float* __restrict__ w_qkv,
                              __bf16* __restrict__ Wbf) {
  int idx = blockIdx.x * 256 + threadIdx.x;
  if (idx >= 320 * 384) return;
  int co = idx / 384, kk = idx - co * 384;
  float v = (co < 64) ? w_conv[co * 384 + kk] : w_qkv[(co - 64) * 384 + kk];
  Wbf[idx] = (__bf16)v;
}

// ---------------------------------------------------------------------------
// Pack V (qkv channels 128..255) to bf16: Vbf[b][c][j], c = h*16+dv.
// ---------------------------------------------------------------------------
__global__ void pack_v_kernel(const float* __restrict__ qkv,
                              __bf16* __restrict__ Vbf) {
  int idx = blockIdx.x * 256 + threadIdx.x;
  if (idx >= N_B * 128 * L_SEQ) return;
  int b = idx >> 17;               // / (128*1024)
  int r = idx & (128 * 1024 - 1);
  Vbf[idx] = (__bf16)qkv[((size_t)b * 256 + 128) * L_SEQ + r];
}

// ---------------------------------------------------------------------------
// Fused conv: out[b][0..63][l] (direct) and qkv[b][0..255][l] (workspace).
// GEMM: [320 x 384] * im2col[384 x 64] per block (one b, 64 l-positions).
// ---------------------------------------------------------------------------
__global__ __launch_bounds__(256) void conv_wmma_kernel(
    const float* __restrict__ x, const __bf16* __restrict__ Wbf,
    const float* __restrict__ b_conv, const float* __restrict__ b_qkv,
    float* __restrict__ out, float* __restrict__ qkv) {
  __shared__ __attribute__((aligned(64))) __bf16 patch[64 * 384];  // [col][kk]

  int b  = blockIdx.x >> 4;
  int l0 = (blockIdx.x & 15) * 64;
  int tid = threadIdx.x, wave = tid >> 5, lane = tid & 31;
  int n = lane & 15, hi = lane & 16;

  // im2col staging (bf16), zero-padded at sequence edges
  for (int idx = tid; idx < 64 * 384; idx += 256) {
    int col = idx & 63, kk = idx >> 6;
    int ci = kk / 3, k = kk - ci * 3;
    int gl = l0 + col + k - 1;
    float v = (gl >= 0 && gl < L_SEQ)
                  ? x[((size_t)b * C_IN + ci) * L_SEQ + gl] : 0.0f;
    patch[col * 384 + kk] = (__bf16)v;
  }
  __syncthreads();

  int halfrow = hi >> 1;  // 0 or 8
  for (int mt = wave; mt < 20; mt += 8) {
    v8f acc[4] = {v8f{}, v8f{}, v8f{}, v8f{}};
    int co_row = mt * 16 + n;  // A-layout: row = lane&15 for all lanes
#pragma unroll
    for (int ks = 0; ks < 12; ks++) {
      // A fragment (16x32 bf16): lanes<16 hold K 0..15, upper lanes K 16..31
      const __bf16* wr = Wbf + (size_t)co_row * 384 + ks * 32 + hi;
      v8bf alo = *(const v8bf*)wr;
      v8bf ahi = *(const v8bf*)(wr + 8);
      v16bf a;
#pragma unroll
      for (int e = 0; e < 8; e++) { a[e] = alo[e]; a[e + 8] = ahi[e]; }
#pragma unroll
      for (int nt = 0; nt < 4; nt++) {
        // B fragment (32x16): col = lane&15, K contiguous; upper lanes K+16
        const __bf16* pb = &patch[(nt * 16 + n) * 384 + ks * 32 + hi];
        acc[nt] = wmma_bf16(a, *(const v16bf*)pb, acc[nt]);
      }
    }
    // C layout: row = v + 8*(lane>=16), col = lane&15
#pragma unroll
    for (int nt = 0; nt < 4; nt++) {
#pragma unroll
      for (int v = 0; v < 8; v++) {
        int co = mt * 16 + v + halfrow;
        int l  = l0 + nt * 16 + n;
        float val = acc[nt][v];
        if (co < 64)
          out[((size_t)b * 128 + co) * L_SEQ + l] = val + b_conv[co];
        else
          qkv[((size_t)b * 256 + (co - 64)) * L_SEQ + l] = val + b_qkv[co - 64];
      }
    }
  }
}

// ---------------------------------------------------------------------------
// Relative-logit 16x16 tile: R[ii,c] = Q[ii,:] . relw[mbase+c,:]  (K=8)
// ---------------------------------------------------------------------------
__device__ __forceinline__ v8f rel_tile(const float* __restrict__ relw,
                                        int mbase, int n, int off, v2f aq0,
                                        v2f aq1) {
  int m = mbase + n;
  m = (m > 2046) ? 2046 : m;  // only unused gather columns can exceed range
  const float* wr = relw + m * 8 + off;
  v8f r = {};
  r = wmma_f32_k4(aq0, mk2(wr[0], wr[1]), r);
  r = wmma_f32_k4(aq1, mk2(wr[4], wr[5]), r);
  return r;
}

// ---------------------------------------------------------------------------
// Fused attention: block = (b, h, 64-row group); 4 waves x 16 Q-rows each.
// Streams 32 key columns per iteration with online softmax (flash style).
// ---------------------------------------------------------------------------
__global__ __launch_bounds__(128) void attn_kernel(
    const float* __restrict__ qkv, const __bf16* __restrict__ Vbf,
    const float* __restrict__ relw, float* __restrict__ attnP) {
  __shared__ __attribute__((aligned(64))) float  Klds[L_SEQ * DKH_];  // 32KB
  __shared__ __attribute__((aligned(64))) __bf16 Pbuf[4][16 * 32];    // 4KB

  int bh  = blockIdx.x >> 4;
  int it0 = (blockIdx.x & 15) * 64;
  int b = bh >> 3, h = bh & 7;
  int tid = threadIdx.x, wave = tid >> 5, lane = tid & 31;
  int n = lane & 15, hi = lane & 16;
  int off = hi >> 3;       // f32 K4 fragment: K offset 0 or 2
  int halfrow = hi >> 1;   // 0 or 8

  const float* qkv_b = qkv + (size_t)b * 256 * L_SEQ;

  // Stage K transposed: Klds[j][d] = qkv[b][64 + h*8 + d][j]
  for (int idx = tid; idx < L_SEQ * DKH_; idx += 128) {
    int d = idx >> 10, j = idx & (L_SEQ - 1);
    Klds[j * DKH_ + d] = qkv_b[(64 + h * DKH_ + d) * L_SEQ + j];
  }
  __syncthreads();

  // Q A-fragments (16x4 f32 layout): a = {Q[m][kb+off], Q[m][kb+off+1]}
  int it = it0 + wave * 16;
  const float scale = 0.35355339059327379f;  // dkh^-0.5, folded into Q
  v2f aq0, aq1;
  {
    const float* qb = qkv_b + (size_t)(h * DKH_) * L_SEQ + (it + n);
    aq0 = mk2(qb[(0 + off) * L_SEQ], qb[(1 + off) * L_SEQ]) * scale;
    aq1 = mk2(qb[(4 + off) * L_SEQ], qb[(5 + off) * L_SEQ]) * scale;
  }

  const __bf16* vb = Vbf + ((size_t)bh * DVH_ + n) * L_SEQ;
  __bf16* pb = &Pbuf[wave][0];

  v8f acc = {};
  float mr[8], lr[8];
#pragma unroll
  for (int v = 0; v < 8; v++) { mr[v] = -3.0e38f; lr[v] = 0.0f; }

  for (int jt = 0; jt < L_SEQ; jt += 32) {
    // ---- S = Q*K^T, two 16x16 tiles (K=8 via two 16x16x4 f32 WMMAs) ----
    v8f s0 = {}, s1 = {};
    {
      const float* kr = &Klds[(jt + n) * DKH_ + off];
      s0 = wmma_f32_k4(aq0, mk2(kr[0], kr[1]), s0);
      s0 = wmma_f32_k4(aq1, mk2(kr[4], kr[5]), s0);
      const float* k2 = &Klds[(jt + 16 + n) * DKH_ + off];
      s1 = wmma_f32_k4(aq0, mk2(k2[0], k2[1]), s1);
      s1 = wmma_f32_k4(aq1, mk2(k2[4], k2[5]), s1);
    }

    // ---- rel logits: 3 overlapping column tiles, diagonal-gathered ----
    int mbase = jt - it + 1008;  // m(i,j) = j-i+1023 ; c = n - M + 15
    v8f r0 = rel_tile(relw, mbase,      n, off, aq0, aq1);
    v8f r1 = rel_tile(relw, mbase + 16, n, off, aq0, aq1);
    v8f r2 = rel_tile(relw, mbase + 32, n, off, aq0, aq1);
#pragma unroll
    for (int v = 0; v < 8; v++) {
      int M = v + halfrow;
      int c = n - M + 15;              // 0..30
      int srcl = (c & 15) | hi;        // same lane-half holds same row
      float a0 = bperm(srcl, r0[v]);
      float b0 = bperm(srcl, r1[v]);
      s0[v] += (c < 16) ? a0 : b0;
      float a1 = bperm(srcl, r1[v]);
      float b1 = bperm(srcl, r2[v]);
      s1[v] += (c < 16) ? a1 : b1;
    }

    // ---- online softmax (row = 16 lanes of one half, per C-VGPR) ----
#pragma unroll
    for (int v = 0; v < 8; v++) {
      float t = fmaxf(s0[v], s1[v]);
      t = fmaxf(t, __shfl_xor(t, 1));
      t = fmaxf(t, __shfl_xor(t, 2));
      t = fmaxf(t, __shfl_xor(t, 4));
      t = fmaxf(t, __shfl_xor(t, 8));
      float mn = fmaxf(mr[v], t);
      float al = __expf(mr[v] - mn);
      mr[v] = mn;
      float p0 = __expf(s0[v] - mn);
      float p1 = __expf(s1[v] - mn);
      float rs = p0 + p1;
      rs += __shfl_xor(rs, 1);
      rs += __shfl_xor(rs, 2);
      rs += __shfl_xor(rs, 4);
      rs += __shfl_xor(rs, 8);
      lr[v] = lr[v] * al + rs;
      acc[v] *= al;
      int row = v + halfrow;           // C-layout -> [row][col] in LDS
      pb[row * 32 + n]      = (__bf16)p0;
      pb[row * 32 + 16 + n] = (__bf16)p1;
    }
    __asm__ volatile("" ::: "memory");  // same-wave DS ops are in order

    // ---- acc += P * V  (bf16 16x16x32 WMMA) ----
    {
      const __bf16* pr = &pb[n * 32 + hi];       // A: row=lane&15
      v8bf plo = *(const v8bf*)pr;               // K 0..7  (+16 upper lanes)
      v8bf phi = *(const v8bf*)(pr + 8);         // K 8..15 (+16 upper lanes)
      v16bf a;
#pragma unroll
      for (int e = 0; e < 8; e++) { a[e] = plo[e]; a[e + 8] = phi[e]; }
      v16bf bv = *(const v16bf*)(vb + jt + hi);  // B: col=lane&15, K contig.
      acc = wmma_bf16(a, bv, acc);
    }
    __asm__ volatile("" ::: "memory");
  }

  // epilogue: attnP[b][h][row][dv] = acc/l
  float* op = attnP + (size_t)bh * L_SEQ * DVH_;
#pragma unroll
  for (int v = 0; v < 8; v++) {
    int row = it + v + halfrow;
    op[row * DVH_ + n] = acc[v] / lr[v];
  }
}

// ---------------------------------------------------------------------------
// Final 1x1 conv. The reference's reshape [L,dvh]->[dvh,L] is a flat
// reinterpret, so channel c of the combined-head tensor is simply the flat
// attnP buffer at offset c*1024+l within each batch.
// ---------------------------------------------------------------------------
__global__ __launch_bounds__(256) void final_conv_kernel(
    const float* __restrict__ attnP, const float* __restrict__ w_attn,
    const float* __restrict__ b_attn, float* __restrict__ out) {
  int idx = blockIdx.x * 256 + threadIdx.x;
  if (idx >= N_B * 64 * L_SEQ) return;
  int l = idx & (L_SEQ - 1);
  int o = (idx >> 10) & 63;
  int b = idx >> 16;
  const float* ap = attnP + (size_t)b * 128 * L_SEQ + l;
  const float* wr = w_attn + o * 128;
  float acc = b_attn[o];
#pragma unroll 8
  for (int c = 0; c < 128; c++) acc += wr[c] * ap[(size_t)c * L_SEQ];
  out[((size_t)b * 128 + 64 + o) * L_SEQ + l] = acc;
}

// ---------------------------------------------------------------------------
extern "C" void kernel_launch(void* const* d_in, const int* in_sizes, int n_in,
                              void* d_out, int out_size, void* d_ws,
                              size_t ws_size, hipStream_t stream) {
  (void)in_sizes; (void)n_in; (void)out_size; (void)ws_size;
  const float* x         = (const float*)d_in[0];
  const float* w_conv    = (const float*)d_in[1];
  const float* b_conv    = (const float*)d_in[2];
  const float* w_qkv     = (const float*)d_in[3];
  const float* b_qkv     = (const float*)d_in[4];
  const float* w_attn    = (const float*)d_in[5];
  const float* b_attn    = (const float*)d_in[6];
  const float* key_rel_w = (const float*)d_in[7];
  float* out = (float*)d_out;

  char* ws = (char*)d_ws;
  float*  qkv   = (float*)(ws);                    // 16*256*1024*4 = 16 MB
  float*  attnP = (float*)(ws + (16u << 20));      // 16*8*1024*16*4 = 8 MB
  __bf16* Vbf   = (__bf16*)(ws + (24u << 20));     // 16*128*1024*2 = 4 MB
  __bf16* Wbf   = (__bf16*)(ws + (28u << 20));     // 320*384*2 ~= 240 KB

  pack_w_kernel<<<(320 * 384 + 255) / 256, 256, 0, stream>>>(w_conv, w_qkv, Wbf);
  conv_wmma_kernel<<<N_B * 16, 256, 0, stream>>>(x, Wbf, b_conv, b_qkv, out, qkv);
  pack_v_kernel<<<(N_B * 128 * L_SEQ) / 256, 256, 0, stream>>>(qkv, Vbf);
  attn_kernel<<<N_B * N_H * 16, 128, 0, stream>>>(qkv, Vbf, key_rel_w, attnP);
  final_conv_kernel<<<(N_B * 64 * L_SEQ) / 256, 256, 0, stream>>>(attnP, w_attn,
                                                                  b_attn, out);
}
